// TemporalGraphConv_28406913696396
// MI455X (gfx1250) — compile-verified, matrix operand order checked by reading, not run on hardware
//
#include <hip/hip_runtime.h>

// ---------------------------------------------------------------------------
// TemporalGraphConv for MI455X (gfx1250): bf16 WMMA GEMMs with fragment-ready
// ("packed") LDS operand layouts (fragment load = 2x ds_load_b128, conflict-
// free), async global->LDS B-panel fills (ASYNCcnt), vectorized f32 A-packing,
// L2-resident edge scatter, last-timestep-only output path.
// ---------------------------------------------------------------------------

#define TT 8
#define NN 50000
#define DD 128
#define HH 128
#define MM 64
#define EE 800000

typedef __bf16 bf16;
typedef __attribute__((ext_vector_type(16))) __bf16 v16bf;
typedef __attribute__((ext_vector_type(8)))  __bf16 v8bf;
typedef __attribute__((ext_vector_type(8)))  float  v8f;

// Packed fragment layout (per 16x32 A-tile or 32x16 B-tile = 512 bf16):
//   element index = frag*512 + half*256 + lane*8 + e   (e in 0..7)
// mapping to matrix coords:
//   A-tile: m = (lane&15),  k = ((lane>>4)<<3) + (half<<4) + e
//   B-tile: n = (lane&15),  k = ((lane>>4)<<3) + (half<<4) + e
// Per-lane data is 2x16 contiguous bytes at 16B lane stride -> all 64 LDS
// banks walked once per 16 lanes: conflict-free ds_load_b128 pairs.

union FragU {
    struct { v8bf lo, hi; } h;
    v16bf v;
};

__device__ __forceinline__ v16bf frag_packed(const bf16* base, int frag) {
    int lane = threadIdx.x & 31;
    const bf16* p = base + frag * 512 + lane * 8;
    FragU u;
    u.h.lo = *(const v8bf*)p;
    u.h.hi = *(const v8bf*)(p + 256);
    return u.v;
}

#define WMMA_BF16(A, B, C) \
    __builtin_amdgcn_wmma_f32_16x16x32_bf16(false, (A), false, (B), (short)0, (C), false, false)

// Generic LDS pointer -> 32-bit LDS address (aperture low bits are the LDS
// offset per CDNA5 flat-addressing rules).
__device__ __forceinline__ unsigned lds_addr32(const void* p) {
    return (unsigned)(size_t)p;
}

// 16B async copy global -> LDS, tracked by ASYNCcnt (bypasses VGPRs).
__device__ __forceinline__ void async_cp16(unsigned lds, const void* gsrc) {
    asm volatile("global_load_async_to_lds_b128 %0, %1, off"
                 :: "v"(lds), "v"(gsrc) : "memory");
}
__device__ __forceinline__ void async_wait0() {
    asm volatile("s_wait_asynccnt 0" ::: "memory");
}

// Cooperative A-packer: rows [row0, row0+MT*16) of a (possibly concatenated)
// row-major f32 source -> packed bf16 LDS. One thread per 8-element group:
// 2x global_load_b128 + cvt + one ds_store_b128.
template <int MT, int NKC>
__device__ __forceinline__ void pack_a_f32(bf16* dst, const float* g1, int ld1, int split,
                                           const float* g2, int ld2, int row0, int nvalid) {
    const int total8 = MT * NKC * 64;
    for (int i = threadIdx.x; i < total8; i += 256) {
        int lane = i & 31, half = (i >> 5) & 1, frag = i >> 6;
        int kc32 = frag % NKC, rt = frag / NKC;
        int m = rt * 16 + (lane & 15);
        int k0 = kc32 * 32 + ((lane >> 4) << 3) + (half << 4);
        int gr = row0 + m;
        v8bf out8;
        if (gr < nvalid) {
            const float* src = (k0 < split) ? (g1 + (size_t)gr * ld1 + k0)
                                            : (g2 + (size_t)gr * ld2 + (k0 - split));
            float4 lo = *(const float4*)src;        // 16B-aligned: k0 % 8 == 0,
            float4 hi = *(const float4*)(src + 4);  // row strides % 64 == 0
            out8[0] = (bf16)lo.x; out8[1] = (bf16)lo.y;
            out8[2] = (bf16)lo.z; out8[3] = (bf16)lo.w;
            out8[4] = (bf16)hi.x; out8[5] = (bf16)hi.y;
            out8[6] = (bf16)hi.z; out8[7] = (bf16)hi.w;
        } else {
#pragma unroll
            for (int e = 0; e < 8; ++e) out8[e] = (bf16)0.f;
        }
        *(v8bf*)(dst + frag * 512 + half * 256 + lane * 8) = out8;
    }
}

// bf16 row-major source -> packed LDS via async global->LDS b128 copies.
template <int MT, int NKC>
__device__ __forceinline__ void pack_a_bf16(bf16* dst, const bf16* g, int ld,
                                            int row0, int nvalid) {
    const int total8 = MT * NKC * 64;
    unsigned ldsbase = lds_addr32(dst);
    for (int i = threadIdx.x; i < total8; i += 256) {
        int lane = i & 31, half = (i >> 5) & 1, frag = i >> 6;
        int kc32 = frag % NKC, rt = frag / NKC;
        int m = rt * 16 + (lane & 15);
        int k0 = kc32 * 32 + ((lane >> 4) << 3) + (half << 4);
        int gr = row0 + m;
        int didx = frag * 512 + half * 256 + lane * 8;
        if (gr < nvalid) {
            async_cp16(ldsbase + (unsigned)(didx * 2), g + (size_t)gr * ld + k0);
        } else {
            v8bf z;
#pragma unroll
            for (int e = 0; e < 8; ++e) z[e] = (bf16)0.f;
            *(v8bf*)(dst + didx) = z;
        }
    }
    async_wait0();
}

// Contiguous async copy of a pre-packed global B panel into LDS.
__device__ __forceinline__ void copy_b_lds(bf16* dst, const bf16* src, int nelem) {
    unsigned ldsbase = lds_addr32(dst);
    for (int i = threadIdx.x; i < (nelem >> 3); i += 256)
        async_cp16(ldsbase + (unsigned)(i * 16), src + i * 8);
    async_wait0();
}

// ---------------------------------------------------------------------------
// One-time converters: pack all weight matrices (KxN f32) into fragment-ready
// bf16 B panels (frag = kc32*ntiles + nt), convert edges to int32.
// ---------------------------------------------------------------------------
__device__ __forceinline__ void pack_b_one(bf16* dst, const float* W, int K, int N, int idx) {
    int e = idx & 7, lane = (idx >> 3) & 31, half = (idx >> 8) & 1, frag = idx >> 9;
    int ntiles = N >> 4;
    int nt = frag % ntiles, kc32 = frag / ntiles;
    int n = nt * 16 + (lane & 15);
    int k = kc32 * 32 + ((lane >> 4) << 3) + (half << 4) + e;
    dst[idx] = (bf16)W[(size_t)k * N + n];
}

__global__ void __launch_bounds__(256)
convert_weights_kernel(const float* __restrict__ Wsp, const float* __restrict__ Wmp,
                       const float* __restrict__ Wi,  const float* __restrict__ Wh,
                       const float* __restrict__ Wo,
                       bf16* __restrict__ o_sp, bf16* __restrict__ o_mp,
                       bf16* __restrict__ o_i,  bf16* __restrict__ o_h,
                       bf16* __restrict__ o_o) {
    int i = blockIdx.x * 256 + threadIdx.x;
    if (i < DD * HH)        pack_b_one(o_sp, Wsp, DD, HH, i);        // 128x128
    if (i < 2 * HH * MM)    pack_b_one(o_mp, Wmp, 2 * HH, MM, i);    // 256x64
    if (i < MM * 3 * MM)    pack_b_one(o_i,  Wi,  MM, 3 * MM, i);    // 64x192
    if (i < MM * 3 * MM)    pack_b_one(o_h,  Wh,  MM, 3 * MM, i);    // 64x192
    if (i < (HH + MM) * HH) pack_b_one(o_o,  Wo,  HH + MM, HH, i);   // 192x128
}

__global__ void __launch_bounds__(256)
convert_edges_kernel(const long long* __restrict__ ei, int* __restrict__ e32) {
    int i = blockIdx.x * 256 + threadIdx.x;
    if (i < 2 * EE) e32[i] = (int)ei[i];
}

// ---------------------------------------------------------------------------
// s = relu(X_t @ W_sp + b_sp)   (N x 128)(128 x 128)
// 64 rows/block; waves own 16x64 strips. A: 4x4 frags (16KB), B: 32KB.
// ---------------------------------------------------------------------------
__global__ void __launch_bounds__(256)
sp_gemm_kernel(const float* __restrict__ X, const bf16* __restrict__ WspP,
               const float* __restrict__ bsp, float* __restrict__ S) {
    __shared__ __align__(16) bf16 sA[4 * 4 * 512];
    __shared__ __align__(16) bf16 sB[DD * HH];
    int row0 = blockIdx.x * 64;
    copy_b_lds(sB, WspP, DD * HH);
    pack_a_f32<4, 4>(sA, X, DD, DD, X, DD, row0, NN);
    __syncthreads();

    int w = threadIdx.x >> 5;
    int rt = w >> 1;           // 0..3
    int ct0 = (w & 1) * 4;     // 0 or 4 (of 8 col tiles)
    v8f acc[4] = {};
#pragma unroll
    for (int kc = 0; kc < 4; ++kc) {
        v16bf a = frag_packed(sA, rt * 4 + kc);
#pragma unroll
        for (int j = 0; j < 4; ++j) {
            v16bf b = frag_packed(sB, kc * 8 + ct0 + j);
            acc[j] = WMMA_BF16(a, b, acc[j]);
        }
    }
    int lane = threadIdx.x & 31;
    int mo = (lane >> 4) << 3, n = lane & 15;
#pragma unroll
    for (int j = 0; j < 4; ++j) {
        int col = (ct0 + j) * 16 + n;
        float bias = bsp[col];
#pragma unroll
        for (int r = 0; r < 8; ++r) {
            int gr = row0 + rt * 16 + mo + r;
            if (gr < NN) {
                float v = acc[j][r] + bias;
                S[(size_t)gr * HH + col] = v > 0.f ? v : 0.f;
            }
        }
    }
}

// ---------------------------------------------------------------------------
// msg[tgt] += s[src] : thread = (edge, 4-float chunk), float4 gather +
// 4x global_atomic_add_f32 (all L2-resident).
// ---------------------------------------------------------------------------
__global__ void __launch_bounds__(256)
scatter_kernel(const int* __restrict__ src, const int* __restrict__ tgt,
               const float* __restrict__ S, float* __restrict__ MSG) {
    long long idx = (long long)blockIdx.x * 256 + threadIdx.x;
    int e = (int)(idx >> 5);
    if (e >= EE) return;
    int c = ((int)idx & 31) << 2;
    int se = src[e], te = tgt[e];
    float4 v = *(const float4*)(S + (size_t)se * HH + c);
    float* p = MSG + (size_t)te * HH + c;
    atomicAdd(p + 0, v.x);
    atomicAdd(p + 1, v.y);
    atomicAdd(p + 2, v.z);
    atomicAdd(p + 3, v.w);
}

// ---------------------------------------------------------------------------
// mi = [s | msg] @ W_mp + b_mp   (N x 256)(256 x 64), output stored bf16
// (only ever a WMMA A operand). 32 rows/block; 8 waves = 2x4 tiles.
// ---------------------------------------------------------------------------
__global__ void __launch_bounds__(256)
mi_gemm_kernel(const float* __restrict__ S, const float* __restrict__ MSG,
               const bf16* __restrict__ WmpP, const float* __restrict__ bmp,
               bf16* __restrict__ MI) {
    __shared__ __align__(16) bf16 sA[2 * 8 * 512];
    __shared__ __align__(16) bf16 sB[2 * HH * MM];
    int row0 = blockIdx.x * 32;
    copy_b_lds(sB, WmpP, 2 * HH * MM);
    pack_a_f32<2, 8>(sA, S, HH, HH, MSG, HH, row0, NN);
    __syncthreads();

    int w = threadIdx.x >> 5;
    int rt = w >> 2, ct = w & 3;
    v8f acc = {};
#pragma unroll
    for (int kc = 0; kc < 8; ++kc) {
        v16bf a = frag_packed(sA, rt * 8 + kc);
        v16bf b = frag_packed(sB, kc * 4 + ct);
        acc = WMMA_BF16(a, b, acc);
    }
    int lane = threadIdx.x & 31;
    int mo = (lane >> 4) << 3, n = lane & 15;
    int col = ct * 16 + n;
    float bias = bmp[col];
#pragma unroll
    for (int r = 0; r < 8; ++r) {
        int gr = row0 + rt * 16 + mo + r;
        if (gr < NN) MI[(size_t)gr * MM + col] = (bf16)(acc[r] + bias);
    }
}

// ---------------------------------------------------------------------------
// GRU cell: each wave owns matching r/z/n column tiles of gi AND gh
// (6 accumulators) so the elementwise update runs in registers on the WMMA
// C layout. h updated in place (unique row/col ownership).
// ---------------------------------------------------------------------------
__global__ void __launch_bounds__(256)
gru_kernel(const bf16* __restrict__ MI, const float* __restrict__ Hst,
           const bf16* __restrict__ WiP, const bf16* __restrict__ WhP,
           const float* __restrict__ bi, const float* __restrict__ bhn,
           float* __restrict__ Hout, float* __restrict__ HFinal, int writeFinal) {
    __shared__ __align__(16) bf16 sMI[2 * 2 * 512];
    __shared__ __align__(16) bf16 sH[2 * 2 * 512];
    __shared__ __align__(16) bf16 sWi[MM * 3 * MM];
    __shared__ __align__(16) bf16 sWh[MM * 3 * MM];
    int row0 = blockIdx.x * 32;
    copy_b_lds(sWi, WiP, MM * 3 * MM);
    copy_b_lds(sWh, WhP, MM * 3 * MM);
    pack_a_bf16<2, 2>(sMI, MI, MM, row0, NN);
    pack_a_f32<2, 2>(sH, Hst, MM, MM, Hst, MM, row0, NN);
    __syncthreads();

    int w = threadIdx.x >> 5;
    int rt = w >> 2, ct = w & 3;              // 2 row tiles x 4 col tiles
    v8f gi[3] = {}, gh[3] = {};
#pragma unroll
    for (int kc = 0; kc < 2; ++kc) {
        v16bf a  = frag_packed(sMI, rt * 2 + kc);
        v16bf ah = frag_packed(sH,  rt * 2 + kc);
#pragma unroll
        for (int p = 0; p < 3; ++p) {         // p: 0=r gate, 1=z gate, 2=n gate
            v16bf b  = frag_packed(sWi, kc * 12 + ct + 4 * p);
            gi[p] = WMMA_BF16(a, b, gi[p]);
            v16bf bh = frag_packed(sWh, kc * 12 + ct + 4 * p);
            gh[p] = WMMA_BF16(ah, bh, gh[p]);
        }
    }
    int lane = threadIdx.x & 31;
    int mo = (lane >> 4) << 3, n = lane & 15;
    int col = ct * 16 + n;
    float b_r = bi[col], b_z = bi[MM + col], b_n = bi[2 * MM + col];
    float bh_n = bhn[col];
#pragma unroll
    for (int r = 0; r < 8; ++r) {
        int gr = row0 + rt * 16 + mo + r;
        if (gr < NN) {
            size_t off = (size_t)gr * MM + col;
            float hprev = Hst[off];
            float rr = 1.f / (1.f + __expf(-(gi[0][r] + b_r + gh[0][r])));
            float zz = 1.f / (1.f + __expf(-(gi[1][r] + b_z + gh[1][r])));
            float nn = tanhf(gi[2][r] + b_n + rr * (gh[2][r] + bh_n));
            float hnew = (1.f - zz) * nn + zz * hprev;
            Hout[off] = hnew;
            if (writeFinal) HFinal[off] = hnew;
        }
    }
}

// ---------------------------------------------------------------------------
// out = [s | h] @ W_out + b_out   (N x 192)(192 x 128); only run at t = T-1.
// ---------------------------------------------------------------------------
__global__ void __launch_bounds__(256)
out_gemm_kernel(const float* __restrict__ S, const float* __restrict__ Hf,
                const bf16* __restrict__ WoP, const float* __restrict__ bo,
                float* __restrict__ OUT) {
    __shared__ __align__(16) bf16 sA[2 * 6 * 512];
    __shared__ __align__(16) bf16 sB[(HH + MM) * HH];
    int row0 = blockIdx.x * 32;
    copy_b_lds(sB, WoP, (HH + MM) * HH);
    pack_a_f32<2, 6>(sA, S, HH, HH, Hf, MM, row0, NN);
    __syncthreads();

    int w = threadIdx.x >> 5;
    int rt = w >> 2;           // 0..1
    int ct0 = (w & 3) * 2;     // 0,2,4,6 (of 8 col tiles)
    v8f acc[2] = {};
#pragma unroll
    for (int kc = 0; kc < 6; ++kc) {
        v16bf a = frag_packed(sA, rt * 6 + kc);
#pragma unroll
        for (int j = 0; j < 2; ++j) {
            v16bf b = frag_packed(sB, kc * 8 + ct0 + j);
            acc[j] = WMMA_BF16(a, b, acc[j]);
        }
    }
    int lane = threadIdx.x & 31;
    int mo = (lane >> 4) << 3, n = lane & 15;
#pragma unroll
    for (int j = 0; j < 2; ++j) {
        int col = (ct0 + j) * 16 + n;
        float bias = bo[col];
#pragma unroll
        for (int r = 0; r < 8; ++r) {
            int gr = row0 + rt * 16 + mo + r;
            if (gr < NN) OUT[(size_t)gr * HH + col] = acc[j][r] + bias;
        }
    }
}

// ---------------------------------------------------------------------------
// Feature-axis conv (window 10, SAME => pad_lo=4):
//   final[n,w] = sum_k out[n, w+k-4] * conv_k[k,0,0] + conv_b[0]
// ---------------------------------------------------------------------------
__global__ void __launch_bounds__(128)
conv_kernel(const float* __restrict__ OUT, const float* __restrict__ convk,
            const float* __restrict__ convb, float* __restrict__ FINAL) {
    __shared__ float row[HH + 9];
    __shared__ float kk[10];
    int n = blockIdx.x;
    int tid = threadIdx.x;
    for (int i = tid; i < HH + 9; i += 128) {
        int wcol = i - 4;
        row[i] = (wcol >= 0 && wcol < HH) ? OUT[(size_t)n * HH + wcol] : 0.f;
    }
    if (tid < 10) kk[tid] = convk[tid * HH];   // conv_k[k,0,0], shape (10,1,128)
    __syncthreads();
    float acc = convb[0];
#pragma unroll
    for (int k = 0; k < 10; ++k) acc += row[tid + k] * kk[k];
    FINAL[(size_t)n * HH + tid] = acc;
}

// ---------------------------------------------------------------------------
extern "C" void kernel_launch(void* const* d_in, const int* in_sizes, int n_in,
                              void* d_out, int out_size, void* d_ws, size_t ws_size,
                              hipStream_t stream) {
    (void)in_sizes; (void)n_in; (void)out_size; (void)ws_size;
    const float*      node = (const float*)d_in[0];       // (T,N,D)
    const long long*  ei   = (const long long*)d_in[1];   // (2,E) int64
    const float* Wsp = (const float*)d_in[2];
    const float* bsp = (const float*)d_in[3];
    const float* Wmp = (const float*)d_in[4];
    const float* bmp = (const float*)d_in[5];
    const float* Wi  = (const float*)d_in[6];
    const float* bi  = (const float*)d_in[7];
    const float* Wh  = (const float*)d_in[8];
    const float* bhn = (const float*)d_in[9];
    const float* Wo  = (const float*)d_in[10];
    const float* bo  = (const float*)d_in[11];
    const float* ck  = (const float*)d_in[12];
    const float* cb  = (const float*)d_in[13];

    char* ws = (char*)d_ws;
    float* S    = (float*)(ws + 0);                // N*H f32   25.6 MB
    float* MSG  = (float*)(ws + 25600000);         // N*H f32   25.6 MB
    float* H    = (float*)(ws + 51200000);         // N*M f32   12.8 MB
    float* OUT  = (float*)(ws + 64000000);         // N*H f32   25.6 MB
    bf16*  MI   = (bf16*) (ws + 89600000);         // N*M bf16   6.4 MB
    bf16*  WspP = (bf16*) (ws + 96000000);         // 16384 packed
    bf16*  WmpP = (bf16*) (ws + 96032768);         // 16384 packed
    bf16*  WiP  = (bf16*) (ws + 96065536);         // 12288 packed
    bf16*  WhP  = (bf16*) (ws + 96090112);         // 12288 packed
    bf16*  WoP  = (bf16*) (ws + 96114688);         // 24576 packed
    int*   E32  = (int*)  (ws + 96163840);         // 2*E int32  6.4 MB

    float* dout_final = (float*)d_out;                       // (N,H)
    float* dout_hfin  = (float*)d_out + (size_t)NN * HH;     // (N,M)

    // once per launch: packed bf16 weight panels, int32 edges, h0 = 0
    convert_weights_kernel<<<96, 256, 0, stream>>>(Wsp, Wmp, Wi, Wh, Wo,
                                                   WspP, WmpP, WiP, WhP, WoP);
    convert_edges_kernel<<<(2 * EE + 255) / 256, 256, 0, stream>>>(ei, E32);
    hipMemsetAsync(H, 0, (size_t)NN * MM * sizeof(float), stream);

    const int spBlocks  = (NN + 63) / 64;    // 782
    const int rowBlocks = (NN + 31) / 32;    // 1563
    const int scBlocks  = (EE * 32) / 256;   // 100000

    for (int t = 0; t < TT; ++t) {
        hipMemsetAsync(MSG, 0, (size_t)NN * HH * sizeof(float), stream);
        sp_gemm_kernel<<<spBlocks, 256, 0, stream>>>(
            node + (size_t)t * NN * DD, WspP, bsp, S);
        scatter_kernel<<<scBlocks, 256, 0, stream>>>(E32, E32 + EE, S, MSG);
        mi_gemm_kernel<<<rowBlocks, 256, 0, stream>>>(S, MSG, WmpP, bmp, MI);
        gru_kernel<<<rowBlocks, 256, 0, stream>>>(
            MI, H, WiP, WhP, bi, bhn, H, dout_hfin, (t == TT - 1) ? 1 : 0);
    }
    // only the last timestep's `out` feeds `final`
    out_gemm_kernel<<<rowBlocks, 256, 0, stream>>>(S, H, WoP, bo, OUT);
    conv_kernel<<<NN, 128, 0, stream>>>(OUT, ck, cb, dout_final);
}
